// HeadwiseEdgeBiasedViewEncoderV5_47553877902011
// MI455X (gfx1250) — matrix operand order repaired and checked by reading, not compiled, not run on hardware
//
#include <hip/hip_runtime.h>

#define B_N   8192
#define K_N   64
#define DIN   256
#define H_N   8
#define DH_N  32
#define OBS_N 32
#define REP_N 256
#define EPS_W   1e-6f
#define LN_EPS  1e-5f

typedef __attribute__((ext_vector_type(16))) _Float16 v16h;
typedef __attribute__((ext_vector_type(8)))  _Float16 v8h;
typedef __attribute__((ext_vector_type(4)))  _Float16 v4h;
typedef __attribute__((ext_vector_type(8)))  float    v8f;

// ---- LDS layout (bytes). Union region reused across phases. ----
constexpr int LDA_H = 272;                 // half-stride of As / Hs rows (pad vs 256)
constexpr int LDC_F = 264;                 // float-stride of H1 rows
constexpr int LDK_H = 34;                  // half-stride of K/V rows (pad vs 32)
constexpr int SZ_AS    = 64 * LDA_H * 2;            // 34816  (x_nei f16)
constexpr int SZ_H1    = 64 * LDC_F * 4;            // 67584  (raw proj f32)
constexpr int SZ_KV    = H_N * 2 * K_N * LDK_H * 2; // 69632  (K+V f16)
constexpr int SZ_UNION = SZ_KV;                     // max(AS,H1,KV)
constexpr int OFF_HS   = SZ_UNION;                  // Hs (LN'd neighbors, f16)
constexpr int OFF_XA   = OFF_HS + SZ_AS;
constexpr int OFF_HRAW = OFF_XA + 1024;
constexpr int OFF_HA   = OFF_HRAW + 1024;
constexpr int OFF_QB   = OFF_HA + 1024;
constexpr int OFF_AB   = OFF_QB + 1024;
constexpr int OFF_HOB  = OFF_AB + 2048;
constexpr int OFF_ST   = OFF_HOB + 1024;
constexpr int SMEM_TOTAL = OFF_ST + 32;             // 111648 B

// -------- weight fp32 -> fp16 staging (W_nei, Wk, Wv) --------
__global__ void convert_weights_kernel(const float* __restrict__ Wnei,
                                       const float* __restrict__ Wk,
                                       const float* __restrict__ Wv,
                                       _Float16* __restrict__ wsh) {
  int i = blockIdx.x * 256 + threadIdx.x;           // 65536 threads
  if (i < DIN * REP_N) wsh[i] = (_Float16)Wnei[i];
  if (i < H_N * DH_N * DH_N) {
    wsh[DIN * REP_N + i]                    = (_Float16)Wk[i];
    wsh[DIN * REP_N + H_N * DH_N * DH_N + i] = (_Float16)Wv[i];
  }
}

__global__ void __launch_bounds__(256)
enc_kernel(const float* __restrict__ x_anchor, const float* __restrict__ x_nei,
           const float* __restrict__ edge_w,   const float* __restrict__ g_local,
           const float* __restrict__ W_anchor, const float* __restrict__ b_anchor,
           const float* __restrict__ g_anchor, const float* __restrict__ beta_anchor,
           const float* __restrict__ b_nei,    const float* __restrict__ g_nei,
           const float* __restrict__ beta_nei, const float* __restrict__ view_embed,
           const float* __restrict__ Wq,       const float* __restrict__ Wo,
           const float* __restrict__ W_geom,   const float* __restrict__ b_geom,
           const _Float16* __restrict__ Wn_h,  const _Float16* __restrict__ Wk_h,
           const _Float16* __restrict__ Wv_h,
           float* __restrict__ head_out,       float* __restrict__ attn_out) {
  extern __shared__ __attribute__((aligned(128))) char smem[];
  _Float16* As = (_Float16*)(smem);                 // union
  float*    H1 = (float*)(smem);                    // union
  _Float16* Kb = (_Float16*)(smem);                 // union
  _Float16* Vb = (_Float16*)(smem + H_N * K_N * LDK_H * 2);
  _Float16* Hs = (_Float16*)(smem + OFF_HS);
  float* xa   = (float*)(smem + OFF_XA);
  float* hraw = (float*)(smem + OFF_HRAW);
  float* ha   = (float*)(smem + OFF_HA);
  float* qb   = (float*)(smem + OFF_QB);
  float* ab   = (float*)(smem + OFF_AB);
  float* hob  = (float*)(smem + OFF_HOB);
  float* st   = (float*)(smem + OFF_ST);

  const int b    = blockIdx.x;
  const int t    = threadIdx.x;
  const int w    = t >> 5;        // wave id == head id (8 waves of 32)
  const int lane = t & 31;
  const int lr   = lane & 15;
  const int kh   = lane >> 4;

  // ---- Phase 1: stage anchor row + neighbor matrix (f32 -> f16) into LDS ----
  xa[t] = x_anchor[(size_t)b * DIN + t];
  {
    const float4* src = (const float4*)(x_nei + (size_t)b * K_N * DIN);
    #pragma unroll
    for (int it = 0; it < 16; ++it) {
      int v = it * 256 + t;                       // 4096 float4's
      float4 f = src[v];
      int e = v * 4, row = e >> 8, col = e & 255;
      v4h hv; hv.x = (_Float16)f.x; hv.y = (_Float16)f.y;
              hv.z = (_Float16)f.z; hv.w = (_Float16)f.w;
      *(v4h*)(&As[row * LDA_H + col]) = hv;
    }
  }
  __syncthreads();

  // ---- Phase 2: anchor projection GEMV (thread = output channel) ----
  {
    float acc = b_anchor[t];
    const float* wrow = W_anchor + (size_t)t * DIN;
    for (int i = 0; i < DIN; ++i) acc += xa[i] * wrow[i];
    hraw[t] = acc;
  }
  __syncthreads();
  if (t == 0) {                                    // anchor LN stats (tiny)
    float m = 0.f; for (int i = 0; i < REP_N; ++i) m += hraw[i];
    m *= (1.f / REP_N);
    float v = 0.f; for (int i = 0; i < REP_N; ++i) { float d = hraw[i] - m; v += d * d; }
    v *= (1.f / REP_N);
    st[0] = m; st[1] = rsqrtf(v + LN_EPS);
  }
  __syncthreads();
  ha[t] = (hraw[t] - st[0]) * st[1] * g_anchor[t] + beta_anchor[t] + view_embed[t & 31];
  __syncthreads();

  // ---- Phase 4: neighbor projection 64x256x256 GEMM via WMMA ----
  // wave w owns output columns [w*32, w*32+32); 4 M-tiles x 2 N-tiles x 8 k-steps
  v8f acc[4][2] = {};
  for (int kk = 0; kk < 8; ++kk) {
    const int k0 = kk * 32;
    v16h a[4];
    #pragma unroll
    for (int mt = 0; mt < 4; ++mt) {               // A fragment: two 8-half chunks
      const _Float16* rp = &As[(mt * 16 + lr) * LDA_H + k0 + kh * 16];
      v8h lo = *(const v8h*)rp;
      v8h hi = *(const v8h*)(rp + 8);
      a[mt] = __builtin_shufflevector(lo, hi, 0,1,2,3,4,5,6,7,8,9,10,11,12,13,14,15);
    }
    #pragma unroll
    for (int nt = 0; nt < 2; ++nt) {               // B fragment: contiguous 16 halves of W row
      const int ng = w * 32 + nt * 16 + lr;
      v16h bf = *(const v16h*)(&Wn_h[(size_t)ng * DIN + k0 + kh * 16]);
      #pragma unroll
      for (int mt = 0; mt < 4; ++mt)
        acc[mt][nt] = __builtin_amdgcn_wmma_f32_16x16x32_f16(
            false, a[mt], false, bf, (short)0, acc[mt][nt], false, false);
    }
  }
  __syncthreads();                                 // As fully read; union becomes H1
  #pragma unroll
  for (int nt = 0; nt < 2; ++nt) {
    const int ng = w * 32 + nt * 16 + lr;
    const float bias = b_nei[ng];
    #pragma unroll
    for (int mt = 0; mt < 4; ++mt) {
      const int mbase = mt * 16 + kh * 8;
      #pragma unroll
      for (int j = 0; j < 8; ++j)
        H1[(mbase + j) * LDC_F + ng] = acc[mt][nt][j] + bias;
    }
  }
  __syncthreads();

  // ---- Phase 5: per-row LayerNorm of neighbors -> Hs (f16) ----
  if (t < 64) {
    const int r = t;
    float m = 0.f;
    for (int c = 0; c < REP_N; ++c) m += H1[r * LDC_F + c];
    m *= (1.f / REP_N);
    float v = 0.f;
    for (int c = 0; c < REP_N; ++c) { float d = H1[r * LDC_F + c] - m; v += d * d; }
    v *= (1.f / REP_N);
    const float rs = rsqrtf(v + LN_EPS);
    for (int c = 0; c < REP_N; ++c)
      Hs[r * LDA_H + c] = (_Float16)((H1[r * LDC_F + c] - m) * rs * g_nei[c] + beta_nei[c]);
  }
  __syncthreads();

  // ---- Phase 6: q (per-head GEMV) + geom scale (softplus) ----
  {
    float s = 0.f;
    const float* wq = Wq + (w * DH_N + lane) * DH_N;
    const float* hh = ha + w * DH_N;
    for (int e = 0; e < DH_N; ++e) s += wq[e] * hh[e];
    qb[w * DH_N + lane] = s;
  }
  float gsc;
  {
    float p = g_local[(size_t)b * OBS_N + lane] * W_geom[w * OBS_N + lane];
    for (int off = 16; off > 0; off >>= 1) p += __shfl_xor(p, off, 32);
    p += b_geom[w];
    gsc = (p > 20.f) ? p : log1pf(expf(p));
  }
  __syncthreads();                                 // H1 dead; union becomes Kb/Vb

  // ---- Phase 7: per-head K/V 64x32x32 GEMMs via WMMA (wave w = head w) ----
  {
    v16h ahf[4];
    #pragma unroll
    for (int mt = 0; mt < 4; ++mt) {
      const _Float16* rp = &Hs[(mt * 16 + lr) * LDA_H + w * DH_N + kh * 16];
      v8h lo = *(const v8h*)rp;
      v8h hi = *(const v8h*)(rp + 8);
      ahf[mt] = __builtin_shufflevector(lo, hi, 0,1,2,3,4,5,6,7,8,9,10,11,12,13,14,15);
    }
    v8f ka[4][2] = {}; v8f va[4][2] = {};
    #pragma unroll
    for (int nt = 0; nt < 2; ++nt) {
      const int n = nt * 16 + lr;
      v16h bk = *(const v16h*)(&Wk_h[(w * DH_N + n) * DH_N + kh * 16]);
      v16h bv = *(const v16h*)(&Wv_h[(w * DH_N + n) * DH_N + kh * 16]);
      #pragma unroll
      for (int mt = 0; mt < 4; ++mt) {
        ka[mt][nt] = __builtin_amdgcn_wmma_f32_16x16x32_f16(
            false, ahf[mt], false, bk, (short)0, ka[mt][nt], false, false);
        va[mt][nt] = __builtin_amdgcn_wmma_f32_16x16x32_f16(
            false, ahf[mt], false, bv, (short)0, va[mt][nt], false, false);
      }
    }
    _Float16* Kh = Kb + w * K_N * LDK_H;
    _Float16* Vh = Vb + w * K_N * LDK_H;
    #pragma unroll
    for (int nt = 0; nt < 2; ++nt) {
      const int n = nt * 16 + lr;
      #pragma unroll
      for (int mt = 0; mt < 4; ++mt) {
        const int mbase = mt * 16 + kh * 8;
        #pragma unroll
        for (int j = 0; j < 8; ++j) {
          Kh[(mbase + j) * LDK_H + n] = (_Float16)ka[mt][nt][j];
          Vh[(mbase + j) * LDK_H + n] = (_Float16)va[mt][nt][j];
        }
      }
    }
  }
  __syncthreads();

  // ---- Phase 8: scores + edge bias + wave-wide softmax (lane owns n, n+32) ----
  {
    const _Float16* Kh = Kb + w * K_N * LDK_H;
    const int n0 = lane, n1 = lane + 32;
    float s0 = 0.f, s1 = 0.f;
    for (int d = 0; d < DH_N; ++d) {
      const float qd = qb[w * DH_N + d];
      s0 += qd * (float)Kh[n0 * LDK_H + d];
      s1 += qd * (float)Kh[n1 * LDK_H + d];
    }
    const float isq = 0.17677669529663687f;        // 1/sqrt(32)
    const float* ew = edge_w + (size_t)b * K_N;
    s0 = s0 * isq + gsc * logf(ew[n0] + EPS_W);
    s1 = s1 * isq + gsc * logf(ew[n1] + EPS_W);
    float mx = fmaxf(s0, s1);
    for (int off = 16; off > 0; off >>= 1) mx = fmaxf(mx, __shfl_xor(mx, off, 32));
    float e0 = expf(s0 - mx), e1 = expf(s1 - mx);
    float sm = e0 + e1;
    for (int off = 16; off > 0; off >>= 1) sm += __shfl_xor(sm, off, 32);
    const float inv = 1.f / sm;
    const float a0 = e0 * inv, a1 = e1 * inv;
    ab[w * K_N + n0] = a0;
    ab[w * K_N + n1] = a1;
    attn_out[(size_t)b * (H_N * K_N) + w * K_N + n0] = a0;
    attn_out[(size_t)b * (H_N * K_N) + w * K_N + n1] = a1;
  }
  __syncthreads();

  // ---- Phase 9: attn @ V (lane = head dim) ----
  {
    const _Float16* Vh = Vb + w * K_N * LDK_H;
    float s = 0.f;
    for (int n = 0; n < K_N; ++n) s += ab[w * K_N + n] * (float)Vh[n * LDK_H + lane];
    hob[w * DH_N + lane] = s;
  }
  __syncthreads();

  // ---- Phase 10: Wo projection + residual ----
  {
    float s = ha[w * DH_N + lane];
    const float* wo = Wo + (w * DH_N + lane) * DH_N;
    for (int d = 0; d < DH_N; ++d) s += wo[d] * hob[w * DH_N + d];
    head_out[(size_t)b * REP_N + w * DH_N + lane] = s;
  }
}

extern "C" void kernel_launch(void* const* d_in, const int* in_sizes, int n_in,
                              void* d_out, int out_size, void* d_ws, size_t ws_size,
                              hipStream_t stream) {
  (void)in_sizes; (void)n_in; (void)out_size; (void)ws_size;
  const float* x_anchor    = (const float*)d_in[0];
  const float* x_nei       = (const float*)d_in[1];
  const float* edge_w      = (const float*)d_in[2];
  const float* g_local     = (const float*)d_in[3];
  const float* W_anchor    = (const float*)d_in[4];
  const float* b_anchor    = (const float*)d_in[5];
  const float* g_anchor    = (const float*)d_in[6];
  const float* beta_anchor = (const float*)d_in[7];
  const float* W_nei       = (const float*)d_in[8];
  const float* b_nei       = (const float*)d_in[9];
  const float* g_nei       = (const float*)d_in[10];
  const float* beta_nei    = (const float*)d_in[11];
  const float* view_embed  = (const float*)d_in[12];
  const float* Wq          = (const float*)d_in[13];
  const float* Wk          = (const float*)d_in[14];
  const float* Wv          = (const float*)d_in[15];
  const float* Wo          = (const float*)d_in[16];
  const float* W_geom      = (const float*)d_in[17];
  const float* b_geom      = (const float*)d_in[18];

  float* head_out = (float*)d_out;
  float* attn_out = head_out + (size_t)B_N * REP_N;
  _Float16* wsh = (_Float16*)d_ws;
  const _Float16* Wn_h = wsh;
  const _Float16* Wk_h = wsh + DIN * REP_N;
  const _Float16* Wv_h = wsh + DIN * REP_N + H_N * DH_N * DH_N;

  convert_weights_kernel<<<256, 256, 0, stream>>>(W_nei, Wk, Wv, wsh);
  enc_kernel<<<B_N, 256, SMEM_TOTAL, stream>>>(
      x_anchor, x_nei, edge_w, g_local,
      W_anchor, b_anchor, g_anchor, beta_anchor,
      b_nei, g_nei, beta_nei, view_embed,
      Wq, Wo, W_geom, b_geom,
      Wn_h, Wk_h, Wv_h, head_out, attn_out);
}
